// NoParamMultiHeadAttention_7009386627388
// MI455X (gfx1250) — matrix operand order repaired and checked by reading, not compiled, not run on hardware
//
#include <hip/hip_runtime.h>

// Multi-head attention (flash-attention style) for gfx1250 / MI455X.
// bf16 WMMA (v_wmma_f32_16x16x32_bf16) with f32 accumulation, online softmax.
// K and V tiles staged via the Tensor Data Mover (tensor_load_to_lds, double-
// buffered f32), converted once per tile to bf16 LDS by a cooperative phase.
// Softmax row reductions via DPP16 row_ror rotate-reduction (no LDS traffic).

#define NH   12
#define HD   64
#define EMB  768
#define SEQ  4096
#define NB   2
#define QB   128   // query rows per workgroup (8 waves x 16 rows)
#define KT   32    // keys per KV iteration
#define NT   (SEQ / KT)
#define KSTR 72    // LDS row stride (bf16 elems) for K tile, padded
#define VSTR 40    // LDS row stride (bf16 elems) for V^T tile, padded

typedef __attribute__((ext_vector_type(16))) __bf16 v16bf;
typedef __attribute__((ext_vector_type(8)))  float  v8f;
typedef unsigned int  u32x4 __attribute__((ext_vector_type(4)));
typedef int           i32x8 __attribute__((ext_vector_type(8)));
typedef int           i32x4 __attribute__((ext_vector_type(4)));

union BF16Frag { v16bf v; unsigned short u[16]; uint4 q[2]; };
union Pack8    { uint4 u4; unsigned short s[8]; };

__device__ __forceinline__ unsigned short f2bf(float f) {
  union { float f; unsigned u; } c; c.f = f;
  unsigned r = c.u + 0x7FFFu + ((c.u >> 16) & 1u);   // round-to-nearest-even
  return (unsigned short)(r >> 16);
}

// B-fragment (32x16 bf16) slice from bf16 LDS: 16 contiguous bf16.
__device__ __forceinline__ BF16Frag loadB_bf16(const unsigned short* base) {
  BF16Frag f;
  f.q[0] = *(const uint4*)(base);
  f.q[1] = *(const uint4*)(base + 8);
  return f;
}

// DPP16 row_ror rotate within 16-lane rows: ctrl = 0x120 + k.
template <int CTRL>
__device__ __forceinline__ float dpp_ror(float x) {
  return __int_as_float(
      __builtin_amdgcn_update_dpp(0, __float_as_int(x), CTRL, 0xF, 0xF, true));
}
__device__ __forceinline__ float rowmax16(float x) {
  x = fmaxf(x, dpp_ror<0x121>(x));   // ror:1
  x = fmaxf(x, dpp_ror<0x122>(x));   // ror:2
  x = fmaxf(x, dpp_ror<0x124>(x));   // ror:4
  x = fmaxf(x, dpp_ror<0x128>(x));   // ror:8
  return x;
}
__device__ __forceinline__ float rowsum16(float x) {
  x += dpp_ror<0x121>(x);
  x += dpp_ror<0x122>(x);
  x += dpp_ror<0x124>(x);
  x += dpp_ror<0x128>(x);
  return x;
}

// Issue a TDM 2D tile load: KT x HD f32 tile (row stride EMB) -> contiguous LDS.
// D# packing per CDNA5 ISA ch.8 (group0: count/lds/global/type; group1: dims).
__device__ __forceinline__ void tdm_load_tile(const float* gsrc, unsigned lds_byte_off) {
  unsigned long long ga = (unsigned long long)(uintptr_t)gsrc;
  u32x4 g0;
  g0[0] = 1u;                                     // count=1, user descriptor
  g0[1] = lds_byte_off;                           // lds_addr      (bits 63:32)
  g0[2] = (unsigned)(ga & 0xFFFFFFFFu);           // global_addr[31:0]
  g0[3] = (unsigned)((ga >> 32) & 0x01FFFFFFu)    // global_addr[56:32]
        | (2u << 30);                             // type=2 (bits 127:126)
  unsigned w[8] = {0, 0, 0, 0, 0, 0, 0, 0};
  w[0]  = 2u << 16;                               // data_size=2 (4 bytes), wg_mask=0
  w[1] |= (HD & 0xFFFFu) << 16;                   // tensor_dim0[15:0]   (bits 63:48)
  w[2] |= (HD >> 16) & 0xFFFFu;                   // tensor_dim0[31:16]  (bits 79:64)
  w[2] |= (KT & 0xFFFFu) << 16;                   // tensor_dim1[15:0]   (bits 95:80)
  w[3] |= (KT >> 16) & 0xFFFFu;                   // tensor_dim1[31:16]  (bits 111:96)
  w[3] |= (unsigned)HD << 16;                     // tile_dim0           (bits 127:112)
  w[4] |= (unsigned)KT;                           // tile_dim1           (bits 143:128)
  w[5]  = (unsigned)EMB;                          // tensor_dim0_stride  (bits 191:160)
  i32x8 g1;
  g1[0] = (int)w[0]; g1[1] = (int)w[1]; g1[2] = (int)w[2]; g1[3] = (int)w[3];
  g1[4] = (int)w[4]; g1[5] = (int)w[5]; g1[6] = (int)w[6]; g1[7] = (int)w[7];
  i32x4 z4 = {0, 0, 0, 0};
  i32x8 z8 = {0, 0, 0, 0, 0, 0, 0, 0};
  __builtin_amdgcn_tensor_load_to_lds(g0, g1, z4, z4, z8, 0);
}

__global__ __launch_bounds__(256)
void mha_flash_bf16_wmma(const float* __restrict__ Qg,
                         const float* __restrict__ Vg,
                         const float* __restrict__ Kg,
                         const int*   __restrict__ Mg,
                         float*       __restrict__ Og) {
  __shared__ __align__(16) float          KsF[2][KT][HD];   // TDM dest (K), f32, dbl-buffered
  __shared__ __align__(16) float          VsF[2][KT][HD];   // TDM dest (V), f32, dbl-buffered
  __shared__ __align__(16) unsigned short Kbf[KT * KSTR];   // K tile bf16, [key][d]
  __shared__ __align__(16) unsigned short Vt[HD * VSTR];    // V tile bf16 transposed, [d][key]
  __shared__ __align__(16) unsigned short Pw[8 * 16 * 32];  // per-wave P scratch, bf16

  const int n    = blockIdx.z;
  const int h    = blockIdx.y;
  const int qb0  = blockIdx.x * QB;
  const int tid  = threadIdx.x;
  const int wave = tid >> 5;
  const int lane = tid & 31;
  const int col  = lane & 15;   // N index inside 16-wide tiles
  const int hi   = lane >> 4;   // lane half

  const float scale = 0.03608439182435161f;  // 1/sqrt(768)

  // ---- resident Q: two 16x32 bf16 A-fragments (D = 0..31, 32..63) ----
  const int qrow = qb0 + wave * 16 + col;    // A-frag: row = lane%16
  const float* qptr = Qg + ((size_t)(n * SEQ + qrow)) * EMB + h * HD;
  BF16Frag qa[2];
#pragma unroll
  for (int f = 0; f < 2; ++f) {
#pragma unroll
    for (int e = 0; e < 8; ++e) {
      // A layout: elems 0..7 -> K = 16*hi + e ; elems 8..15 -> K = 16*hi + 8 + e
      qa[f].u[e]     = f2bf(qptr[32 * f + 16 * hi + e]);
      qa[f].u[e + 8] = f2bf(qptr[32 * f + 16 * hi + 8 + e]);
    }
  }

  // ---- online softmax state + O accumulator (16x64 f32, C layout x4) ----
  const v8f z8f = {0.f, 0.f, 0.f, 0.f, 0.f, 0.f, 0.f, 0.f};
  v8f o[4] = {z8f, z8f, z8f, z8f};
  float m_r[8], l_r[8];
#pragma unroll
  for (int j = 0; j < 8; ++j) { m_r[j] = -3.0e38f; l_r[j] = 0.f; }

  // convert-phase coordinates: 256 threads x 8 elems = one 32x64 tile
  const int srow = tid >> 3;          // 0..31 (key row)
  const int scol = (tid & 7) * 8;     // 0..56 step 8 (d col)

  // ---- prologue: TDM K+V tile 0 into buffer 0 ----
  if (wave == 0) {
    const size_t tb = ((size_t)(n * SEQ)) * EMB + h * HD;
    tdm_load_tile(Kg + tb, (unsigned)(uintptr_t)&KsF[0][0][0]);
    tdm_load_tile(Vg + tb, (unsigned)(uintptr_t)&VsF[0][0][0]);
  }

  for (int it = 0; it < NT; ++it) {
    const int kvb = it * KT;
    const int buf = it & 1;
    __syncthreads();  // A: all readers of previous LDS contents are done

    // ---- issue TDM for tile it+1 into the other f32 buffer ----
    if (wave == 0) {
      if ((it + 1) < NT) {
        const size_t tb = ((size_t)(n * SEQ + kvb + KT)) * EMB + h * HD;
        tdm_load_tile(Kg + tb, (unsigned)(uintptr_t)&KsF[buf ^ 1][0][0]);
        tdm_load_tile(Vg + tb, (unsigned)(uintptr_t)&VsF[buf ^ 1][0][0]);
        __builtin_amdgcn_s_wait_tensorcnt(2);  // tile it's pair has landed
      } else {
        __builtin_amdgcn_s_wait_tensorcnt(0);
      }
      __builtin_prefetch(Mg + (size_t)n * SEQ + kvb, 0, 3);
    }
    __syncthreads();  // B: f32 tile it visible to all waves

    // ---- cooperative convert: f32 tile -> bf16 K (row-major) + V^T ----
    {
      const float4* kf = (const float4*)&KsF[buf][srow][scol];
      float4 a = kf[0], b = kf[1];
      Pack8 pk;
      pk.s[0] = f2bf(a.x); pk.s[1] = f2bf(a.y); pk.s[2] = f2bf(a.z); pk.s[3] = f2bf(a.w);
      pk.s[4] = f2bf(b.x); pk.s[5] = f2bf(b.y); pk.s[6] = f2bf(b.z); pk.s[7] = f2bf(b.w);
      *(uint4*)&Kbf[srow * KSTR + scol] = pk.u4;

      const float4* vf = (const float4*)&VsF[buf][srow][scol];
      float4 c = vf[0], d = vf[1];
      unsigned short vs[8] = { f2bf(c.x), f2bf(c.y), f2bf(c.z), f2bf(c.w),
                               f2bf(d.x), f2bf(d.y), f2bf(d.z), f2bf(d.w) };
#pragma unroll
      for (int i = 0; i < 8; ++i)
        Vt[(scol + i) * VSTR + srow] = vs[i];
    }
    __syncthreads();  // C: bf16 tiles visible

    // ---- S = Q * K^T : two 16x16 score tiles (keys kvb..+15, +16..+31) ----
    v8f st0 = z8f, st1 = z8f;
    const unsigned short* kr0 = &Kbf[col * KSTR];        // key_local = col
    const unsigned short* kr1 = &Kbf[(16 + col) * KSTR]; // key_local = 16+col
    // B layout: b[e] = B[K = 16*hi + e][N = col]; K^T chunk c at d = 32c..
    BF16Frag b00 = loadB_bf16(kr0 + 16 * hi);
    BF16Frag b01 = loadB_bf16(kr0 + 32 + 16 * hi);
    BF16Frag b10 = loadB_bf16(kr1 + 16 * hi);
    BF16Frag b11 = loadB_bf16(kr1 + 32 + 16 * hi);
    st0 = __builtin_amdgcn_wmma_f32_16x16x32_bf16(false, qa[0].v, false, b00.v, (short)0, st0, false, false);
    st0 = __builtin_amdgcn_wmma_f32_16x16x32_bf16(false, qa[1].v, false, b01.v, (short)0, st0, false, false);
    st1 = __builtin_amdgcn_wmma_f32_16x16x32_bf16(false, qa[0].v, false, b10.v, (short)0, st1, false, false);
    st1 = __builtin_amdgcn_wmma_f32_16x16x32_bf16(false, qa[1].v, false, b11.v, (short)0, st1, false, false);

    // ---- mask as additive bias + online softmax update (DPP reductions) ----
    const float bias0 = (Mg[(size_t)n * SEQ + kvb + col]      == 0) ? -1e20f : 0.f;
    const float bias1 = (Mg[(size_t)n * SEQ + kvb + 16 + col] == 0) ? -1e20f : 0.f;
    float p0[8], p1[8];
#pragma unroll
    for (int j = 0; j < 8; ++j) {
      float s0 = fmaf(st0[j], scale, bias0);
      float s1 = fmaf(st1[j], scale, bias1);
      float mx = rowmax16(fmaxf(s0, s1));          // row-max, pure VALU (DPP16)
      float mnew  = fmaxf(m_r[j], mx);
      float alpha = __expf(m_r[j] - mnew);
      p0[j] = __expf(s0 - mnew);
      p1[j] = __expf(s1 - mnew);
      float rs = rowsum16(p0[j] + p1[j]);          // row-sum, pure VALU (DPP16)
      l_r[j] = l_r[j] * alpha + rs;
      m_r[j] = mnew;
      o[0][j] *= alpha; o[1][j] *= alpha; o[2][j] *= alpha; o[3][j] *= alpha;
    }

    // ---- transpose P (C layout -> A layout) via per-wave LDS round trip ----
    unsigned short* pw = &Pw[wave * 512];
#pragma unroll
    for (int j = 0; j < 8; ++j) {
      pw[(j + 8 * hi) * 32 + col]      = f2bf(p0[j]);   // C layout: row j+8*hi, col
      pw[(j + 8 * hi) * 32 + 16 + col] = f2bf(p1[j]);
    }
    asm volatile("s_wait_dscnt 0" ::: "memory");        // same-wave DS RAW fence
    BF16Frag pa;
    const unsigned short* pr = pw + col * 32;           // A layout: row = col
    pa.q[0] = *(const uint4*)(pr + 16 * hi);
    pa.q[1] = *(const uint4*)(pr + 16 * hi + 8);

    // ---- O += P * V : 4 output column tiles of 16 ----
#pragma unroll
    for (int t = 0; t < 4; ++t) {
      BF16Frag vbf = loadB_bf16(&Vt[(16 * t + col) * VSTR + 16 * hi]);
      o[t] = __builtin_amdgcn_wmma_f32_16x16x32_bf16(false, pa.v, false, vbf.v, (short)0, o[t], false, false);
    }
  }

  // ---- epilogue: normalize and store ----
#pragma unroll
  for (int j = 0; j < 8; ++j) {
    const float inv = 1.0f / l_r[j];
    const int r = qb0 + wave * 16 + j + 8 * hi;
    float* op = Og + ((size_t)(n * SEQ + r)) * EMB + h * HD;
    op[0  + col] = o[0][j] * inv;
    op[16 + col] = o[1][j] * inv;
    op[32 + col] = o[2][j] * inv;
    op[48 + col] = o[3][j] * inv;
  }
}

extern "C" void kernel_launch(void* const* d_in, const int* in_sizes, int n_in,
                              void* d_out, int out_size, void* d_ws, size_t ws_size,
                              hipStream_t stream) {
  const float* queries = (const float*)d_in[0];
  const float* values  = (const float*)d_in[1];
  const float* keys    = (const float*)d_in[2];
  const int*   mask    = (const int*)d_in[3];
  float* out = (float*)d_out;
  (void)in_sizes; (void)n_in; (void)out_size; (void)d_ws; (void)ws_size;

  dim3 grid(SEQ / QB, NH, NB);   // (32, 12, 2)
  dim3 block(256);               // 8 wave32
  mha_flash_bf16_wmma<<<grid, block, 0, stream>>>(queries, values, keys, mask, out);
}
